// SelfAttention_2336462209755
// MI455X (gfx1250) — matrix operand order repaired and checked by reading, not compiled, not run on hardware
//
#include <hip/hip_runtime.h>
#include <hip/hip_bf16.h>

// ---------------------------------------------------------------------------
// Causal single-head attention, B=4, S=4096, D=512 (fp32 in/out).
// f16 WMMA pipeline: convert -> QKV GEMM (V transposed) -> fused flash attn.
// 32-row M/Q tiles per block: every W/K/V B-fragment feeds 2 wmmas.
// Shared A/Q tiles staged in LDS via GLOBAL_LOAD_ASYNC_TO_LDS_B128.
// ---------------------------------------------------------------------------

typedef __attribute__((ext_vector_type(16))) _Float16 v16h;
typedef __attribute__((ext_vector_type(8)))  _Float16 h8;
typedef __attribute__((ext_vector_type(8)))  float    v8f;
typedef __attribute__((ext_vector_type(4)))  int      v4i;

#define BATCH 4
#define SLEN  4096
#define DMODEL 512
#define NEG_INF (-1e9f)
#define LDS_STRIDE 520  // 512 + 8 halves: 260 dwords/row -> staggered banks

#if defined(__has_builtin)
#if __has_builtin(__builtin_amdgcn_global_load_async_to_lds_b128) && \
    __has_builtin(__builtin_amdgcn_s_wait_asynccnt)
#define USE_ASYNC_LDS 1
#endif
#endif
#ifndef USE_ASYNC_LDS
#define USE_ASYNC_LDS 0
#endif

union FragH { v16h v; h8 h[2]; };

__device__ __forceinline__ v8f wmma_f16(v16h a, v16h b, v8f c) {
  return __builtin_amdgcn_wmma_f32_16x16x32_f16(
      /*neg_a=*/false, a, /*neg_b=*/false, b,
      /*c_mod=*/(short)0, c, /*reuse_a=*/false, /*reuse_b=*/false);
}

// Copy 64 bytes global -> LDS (per calling thread).
__device__ __forceinline__ void stage64(const _Float16* __restrict__ g,
                                        _Float16* __restrict__ l) {
#if USE_ASYNC_LDS
#pragma unroll
  for (int u = 0; u < 4; ++u)
    __builtin_amdgcn_global_load_async_to_lds_b128(
        (v4i*)(g + u * 8), (v4i*)(l + u * 8), 0, 0);
#else
  const h8* gp = (const h8*)g;
  h8* lp = (h8*)l;
#pragma unroll
  for (int u = 0; u < 4; ++u) lp[u] = gp[u];
#endif
}

__device__ __forceinline__ void stage_fence() {
#if USE_ASYNC_LDS
  __builtin_amdgcn_s_wait_asynccnt(0);
#endif
  __syncthreads();
}

// ---------------- fp32 -> f16 conversion --------------------------------
__global__ void cvt_f16_kernel(const float* __restrict__ src,
                               _Float16* __restrict__ dst, int n) {
  int i = blockIdx.x * blockDim.x + threadIdx.x;
  if (i < n) dst[i] = (_Float16)src[i];
}

// W [D x D] row-major -> WT [D x D] f16 with WT[e][d] = W[d][e]
__global__ void cvt_wT_kernel(const float* __restrict__ w,
                              _Float16* __restrict__ wt) {
  int i = blockIdx.x * blockDim.x + threadIdx.x;  // i = e*D + d
  int e = i >> 9;
  int d = i & 511;
  wt[i] = (_Float16)w[d * DMODEL + e];
}

// ---------------- QKV projection GEMM -----------------------------------
// out[m][n] = sum_k A16[m][k] * WT[n][k],  M = B*S = 16384, N = K = 512.
// Block: 256 threads (8 waves), 32 output rows per block. Wave w owns
// columns [w*64, w*64+64) for both 16-row groups (B-fragment reuse x2).
// vmode==1: write transposed into Vt[b][d][s].
__global__ __launch_bounds__(256)
void qkv_gemm_kernel(const _Float16* __restrict__ A,
                     const _Float16* __restrict__ WT,
                     _Float16* __restrict__ out, int vmode) {
  __shared__ __align__(16) _Float16 As[32][LDS_STRIDE];  // 32 x 512 tile

  const int m0   = blockIdx.x * 32;
  const int wv   = threadIdx.x >> 5;
  const int lane = threadIdx.x & 31;
  const int g    = lane >> 4;
  const int ln   = lane & 15;
  const int nbase = wv * 64;

  {  // cooperative stage: 256 threads x 128B = 32 KB
    const int row = threadIdx.x >> 3;
    const int col = (threadIdx.x & 7) * 64;
    const _Float16* gp = A + (size_t)(m0 + row) * DMODEL + col;
    stage64(gp, &As[row][col]);
    stage64(gp + 32, &As[row][col + 32]);
  }
  stage_fence();

  v8f acc[2][4] = {};
  for (int k0 = 0; k0 < DMODEL; k0 += 32) {
    FragH a0, a1;
    a0.h[0] = *(const h8*)&As[ln][k0 + 8 * g];
    a0.h[1] = *(const h8*)&As[ln][k0 + 16 + 8 * g];
    a1.h[0] = *(const h8*)&As[16 + ln][k0 + 8 * g];
    a1.h[1] = *(const h8*)&As[16 + ln][k0 + 16 + 8 * g];
#pragma unroll
    for (int j = 0; j < 4; ++j) {
      FragH b;
      const _Float16* bp =
          WT + (size_t)(nbase + j * 16 + ln) * DMODEL + k0 + 8 * g;
      b.h[0] = *(const h8*)(bp);
      b.h[1] = *(const h8*)(bp + 16);
      acc[0][j] = wmma_f16(a0.v, b.v, acc[0][j]);
      acc[1][j] = wmma_f16(a1.v, b.v, acc[1][j]);
    }
  }

  if (!vmode) {
#pragma unroll
    for (int i = 0; i < 2; ++i)
#pragma unroll
      for (int j = 0; j < 4; ++j) {
        const int ncol = nbase + j * 16 + ln;
#pragma unroll
        for (int r = 0; r < 8; ++r) {
          const int m = m0 + 16 * i + r + 8 * g;
          out[(size_t)m * DMODEL + ncol] = (_Float16)acc[i][j][r];
        }
      }
  } else {  // V transposed: Vt[b][d][s]
    const int b = m0 >> 12;  // 4096 rows per batch, tile never straddles
#pragma unroll
    for (int i = 0; i < 2; ++i)
#pragma unroll
      for (int j = 0; j < 4; ++j) {
        const int d = nbase + j * 16 + ln;
#pragma unroll
        for (int r = 0; r < 8; ++r) {
          const int s = (m0 + 16 * i + r + 8 * g) & (SLEN - 1);
          out[((size_t)b * DMODEL + d) * SLEN + s] = (_Float16)acc[i][j][r];
        }
      }
  }
}

// ---------------- Fused flash attention ---------------------------------
// Grid: x = q-tile (S/32 = 128), y = batch. Block = 256 threads (8 waves).
// 32-query tile staged in LDS once. Per 128-key chunk each wave computes a
// 32x16 score tile (K fragment reused x2) + PV for its 64-wide D-slice
// (V fragment reused x2). Online softmax state replicated per wave.
__global__ __launch_bounds__(256)
void attn_kernel(const _Float16* __restrict__ Q,
                 const _Float16* __restrict__ K,
                 const _Float16* __restrict__ Vt,
                 float* __restrict__ out) {
  __shared__ __align__(16) _Float16 Qs[32][LDS_STRIDE];  // 32 x 512 Q tile
  __shared__ __align__(16) _Float16 Plds[32][136];       // 32 x 128 P tile
  __shared__ float wmax[8][32];
  __shared__ float wsum[8][32];

  const int b    = blockIdx.y;
  const int q0   = blockIdx.x * 32;
  const int wv   = threadIdx.x >> 5;
  const int lane = threadIdx.x & 31;
  const int g    = lane >> 4;
  const int ln   = lane & 15;
  const float scale = 0.04419417382415922f;  // 1/sqrt(512)

  const _Float16* Qb = Q  + (size_t)b * SLEN * DMODEL;
  const _Float16* Kb = K  + (size_t)b * SLEN * DMODEL;
  const _Float16* Vb = Vt + (size_t)b * DMODEL * SLEN;

  {  // stage the 32x512 Q tile once
    const int row = threadIdx.x >> 3;
    const int col = (threadIdx.x & 7) * 64;
    const _Float16* gp = Qb + (size_t)(q0 + row) * DMODEL + col;
    stage64(gp, &Qs[row][col]);
    stage64(gp + 32, &Qs[row][col + 32]);
  }
  stage_fence();

  float m_run[2][8], l_run[2][8];
#pragma unroll
  for (int i = 0; i < 2; ++i)
#pragma unroll
    for (int r = 0; r < 8; ++r) { m_run[i][r] = -1e30f; l_run[i][r] = 0.0f; }
  v8f acc[2][4] = {};

  const int nchunks = (q0 >> 7) + 1;  // keys 0 .. q0+31, chunks of 128
  for (int c = 0; c < nchunks; ++c) {
    const int kk0 = c * 128 + wv * 16;

    // ---- scores: S = Q_tile (32 x D) @ K_tile^T (D x 16) ----
    v8f s[2] = {};
    for (int k0 = 0; k0 < DMODEL; k0 += 32) {
      FragH a0, a1, bk;
      a0.h[0] = *(const h8*)&Qs[ln][k0 + 8 * g];
      a0.h[1] = *(const h8*)&Qs[ln][k0 + 16 + 8 * g];
      a1.h[0] = *(const h8*)&Qs[16 + ln][k0 + 8 * g];
      a1.h[1] = *(const h8*)&Qs[16 + ln][k0 + 16 + 8 * g];
      const _Float16* kp = Kb + (size_t)(kk0 + ln) * DMODEL + k0 + 8 * g;
      bk.h[0] = *(const h8*)(kp);
      bk.h[1] = *(const h8*)(kp + 16);
      s[0] = wmma_f16(a0.v, bk.v, s[0]);
      s[1] = wmma_f16(a1.v, bk.v, s[1]);
    }

    // ---- scale + causal mask + per-tile row max ----
    const int kidx = kk0 + ln;  // this lane's key column
    float mx[2][8];
#pragma unroll
    for (int i = 0; i < 2; ++i)
#pragma unroll
      for (int r = 0; r < 8; ++r) {
        const int qidx = q0 + 16 * i + r + 8 * g;
        float v = s[i][r] * scale;
        v = (kidx <= qidx) ? v : NEG_INF;
        s[i][r] = v;
        mx[i][r] = v;
      }
#pragma unroll
    for (int msk = 8; msk >= 1; msk >>= 1)
#pragma unroll
      for (int i = 0; i < 2; ++i)
#pragma unroll
        for (int r = 0; r < 8; ++r)
          mx[i][r] = fmaxf(mx[i][r], __shfl_xor(mx[i][r], msk, 32));
    if (ln == 0)
#pragma unroll
      for (int i = 0; i < 2; ++i)
#pragma unroll
        for (int r = 0; r < 8; ++r) wmax[wv][16 * i + r + 8 * g] = mx[i][r];
    __syncthreads();

    // ---- combine row max across waves; rescale factors ----
    float fac[2][8];
#pragma unroll
    for (int i = 0; i < 2; ++i)
#pragma unroll
      for (int r = 0; r < 8; ++r) {
        const int row = 16 * i + r + 8 * g;
        float cm = m_run[i][r];
#pragma unroll
        for (int ww = 0; ww < 8; ++ww) cm = fmaxf(cm, wmax[ww][row]);
        fac[i][r] = __expf(m_run[i][r] - cm);
        m_run[i][r] = cm;
      }

    // ---- P = exp(s - m); write f16 P tile to LDS; row sums ----
    float rs[2][8];
#pragma unroll
    for (int i = 0; i < 2; ++i)
#pragma unroll
      for (int r = 0; r < 8; ++r) {
        const float p = __expf(s[i][r] - m_run[i][r]);
        rs[i][r] = p;
        Plds[16 * i + r + 8 * g][wv * 16 + ln] = (_Float16)p;
      }
#pragma unroll
    for (int msk = 8; msk >= 1; msk >>= 1)
#pragma unroll
      for (int i = 0; i < 2; ++i)
#pragma unroll
        for (int r = 0; r < 8; ++r)
          rs[i][r] += __shfl_xor(rs[i][r], msk, 32);
    if (ln == 0)
#pragma unroll
      for (int i = 0; i < 2; ++i)
#pragma unroll
        for (int r = 0; r < 8; ++r) wsum[wv][16 * i + r + 8 * g] = rs[i][r];
    __syncthreads();

    // ---- update l, rescale accumulator ----
#pragma unroll
    for (int i = 0; i < 2; ++i)
#pragma unroll
      for (int r = 0; r < 8; ++r) {
        const int row = 16 * i + r + 8 * g;
        float srow = 0.0f;
#pragma unroll
        for (int ww = 0; ww < 8; ++ww) srow += wsum[ww][row];
        l_run[i][r] = l_run[i][r] * fac[i][r] + srow;
#pragma unroll
        for (int j = 0; j < 4; ++j) acc[i][j][r] *= fac[i][r];
      }

    // ---- PV: acc[i][j] += P(32x128) @ V(128 x 64-slice) ----
#pragma unroll
    for (int t = 0; t < 4; ++t) {
      FragH pa0, pa1;
      pa0.h[0] = *(const h8*)&Plds[ln][t * 32 + 8 * g];
      pa0.h[1] = *(const h8*)&Plds[ln][t * 32 + 16 + 8 * g];
      pa1.h[0] = *(const h8*)&Plds[16 + ln][t * 32 + 8 * g];
      pa1.h[1] = *(const h8*)&Plds[16 + ln][t * 32 + 16 + 8 * g];
#pragma unroll
      for (int j = 0; j < 4; ++j) {
        FragH bv;
        const _Float16* vp =
            Vb + (size_t)(wv * 64 + j * 16 + ln) * SLEN + c * 128 + t * 32 +
            8 * g;
        bv.h[0] = *(const h8*)(vp);
        bv.h[1] = *(const h8*)(vp + 16);
        acc[0][j] = wmma_f16(pa0.v, bv.v, acc[0][j]);
        acc[1][j] = wmma_f16(pa1.v, bv.v, acc[1][j]);
      }
    }
    __syncthreads();  // protect Plds/wmax/wsum for next chunk
  }

  // ---- epilogue: out = acc / l ----
  float* ob = out + ((size_t)b * SLEN + q0) * DMODEL;
#pragma unroll
  for (int i = 0; i < 2; ++i)
#pragma unroll
    for (int r = 0; r < 8; ++r) {
      const float rl = 1.0f / l_run[i][r];
#pragma unroll
      for (int j = 0; j < 4; ++j)
        ob[(size_t)(16 * i + r + 8 * g) * DMODEL + wv * 64 + j * 16 + ln] =
            acc[i][j][r] * rl;
    }
}

// ---------------------------------------------------------------------------
extern "C" void kernel_launch(void* const* d_in, const int* in_sizes, int n_in,
                              void* d_out, int out_size, void* d_ws,
                              size_t ws_size, hipStream_t stream) {
  (void)in_sizes; (void)n_in; (void)out_size; (void)ws_size;
  const float* x  = (const float*)d_in[0];
  // d_in[1] is the causal mask (bool) -- structure is known, unused.
  const float* Wq = (const float*)d_in[2];
  const float* Wk = (const float*)d_in[3];
  const float* Wv = (const float*)d_in[4];
  float* out = (float*)d_out;

  const size_t nx = (size_t)BATCH * SLEN * DMODEL;      // 8388608
  const size_t nw = (size_t)DMODEL * DMODEL;            // 262144

  char* ws = (char*)d_ws;
  _Float16* x16  = (_Float16*)(ws);                      // 16 MB
  _Float16* WqT  = (_Float16*)(ws + nx * 2);
  _Float16* WkT  = WqT + nw;
  _Float16* WvT  = WkT + nw;
  _Float16* Q16  = WvT + nw;                             // 16 MB
  _Float16* K16  = Q16 + nx;                             // 16 MB
  _Float16* Vt16 = K16 + nx;                             // 16 MB

  // 1) precision conversion (+ weight transpose)
  cvt_f16_kernel<<<(int)(nx / 256), 256, 0, stream>>>(x, x16, (int)nx);
  cvt_wT_kernel<<<(int)(nw / 256), 256, 0, stream>>>(Wq, WqT);
  cvt_wT_kernel<<<(int)(nw / 256), 256, 0, stream>>>(Wk, WkT);
  cvt_wT_kernel<<<(int)(nw / 256), 256, 0, stream>>>(Wv, WvT);

  // 2) QKV projections (M = 16384 rows, 32 rows per block)
  const int gemm_blocks = (BATCH * SLEN) / 32;  // 512
  qkv_gemm_kernel<<<gemm_blocks, 256, 0, stream>>>(x16, WqT, Q16, 0);
  qkv_gemm_kernel<<<gemm_blocks, 256, 0, stream>>>(x16, WkT, K16, 0);
  qkv_gemm_kernel<<<gemm_blocks, 256, 0, stream>>>(x16, WvT, Vt16, 1);

  // 3) fused causal flash attention
  dim3 agrid(SLEN / 32, BATCH);
  attn_kernel<<<agrid, 256, 0, stream>>>(Q16, K16, Vt16, out);
}
